// GlobalPushPlus_2422361555241
// MI455X (gfx1250) — matrix operand order repaired and checked by reading, not compiled, not run on hardware
//
#include <hip/hip_runtime.h>
#include <hip/hip_bf16.h>

// Problem sizes (fixed by reference)
#define N_ROWS     4096
#define DIM        512
#define N_CENTERS  8192
#define KT_COUNT   (DIM / 32)          // 16 k-tiles of 32

typedef __attribute__((ext_vector_type(16))) __bf16 v16bf;
typedef __attribute__((ext_vector_type(8)))  float  v8f;

// ---------------- workspace layout (bytes) ----------------
static constexpr size_t SZ_A     = (size_t)N_ROWS    * DIM * 2;   // 4 MiB
static constexpr size_t SZ_B     = (size_t)N_CENTERS * DIM * 2;   // 8 MiB
static constexpr size_t OFF_AHI  = 0;
static constexpr size_t OFF_ALO  = OFF_AHI + SZ_A;
static constexpr size_t OFF_BHI  = OFF_ALO + SZ_A;
static constexpr size_t OFF_BLO  = OFF_BHI + SZ_B;
static constexpr size_t OFF_INVF = OFF_BLO + SZ_B;
static constexpr size_t OFF_INVC = OFF_INVF + (size_t)N_ROWS * 4;
static constexpr size_t OFF_POS  = OFF_INVC + (size_t)N_CENTERS * 4;
static constexpr size_t OFF_ACC  = OFF_POS  + (size_t)N_ROWS * 4; // 8B double + 4B u32

// ---------------- helpers ----------------
__device__ __forceinline__ unsigned short bf16_rn(float f) {
    unsigned int u = __float_as_uint(f);
    u += 0x7FFFu + ((u >> 16) & 1u);       // round to nearest even
    return (unsigned short)(u >> 16);
}
__device__ __forceinline__ float bf16_f32(unsigned short h) {
    return __uint_as_float(((unsigned int)h) << 16);
}

// ---------------- kernel: zero accumulators ----------------
__global__ void gpp_init_kernel(double* loss_acc, unsigned int* cnt_acc) {
    *loss_acc = 0.0;
    *cnt_acc  = 0u;
}

// ---------------- kernel: normalize rows + split bf16 + swizzle to WMMA layout ----
// IS_A = 1 : A-matrix layout (16x32 tile, M x K)   for features
// IS_A = 0 : B-matrix layout (32x16 tile, K x N)   for centers (c^T)
template <int IS_A>
__global__ __launch_bounds__(256)
void gpp_norm_swizzle_kernel(const float* __restrict__ src,
                             unsigned short* __restrict__ hi_out,
                             unsigned short* __restrict__ lo_out,
                             float* __restrict__ inv_out) {
    __shared__ float red[256];
    const int r = blockIdx.x;            // row of src  (M for A, N for B)
    const int t = threadIdx.x;
    const float* row = src + (size_t)r * DIM;

    const float x0 = row[t];
    const float x1 = row[t + 256];
    red[t] = x0 * x0 + x1 * x1;
    __syncthreads();
    for (int s = 128; s > 0; s >>= 1) {
        if (t < s) red[t] += red[t + s];
        __syncthreads();
    }
    const float inv = 1.0f / fmaxf(sqrtf(red[0]), 1e-12f);
    if (t == 0) inv_out[r] = inv;

    const int tile_r = r >> 4;
    const int L16    = r & 15;

#pragma unroll
    for (int j = 0; j < 2; ++j) {
        const int   k = t + j * 256;
        const float v = (j ? x1 : x0) * inv;
        const unsigned short h  = bf16_rn(v);
        const unsigned short lo = bf16_rn(v - bf16_f32(h));

        const int kt = k >> 5;
        const int kk = k & 31;
        int lane, e;
        if (IS_A) {   // A 16x32 tile: lanes 0-15 hold K{0..7,16..23}, lanes 16-31 hold K{8..15,24..31}
            lane = L16 + (((kk >> 3) & 1) << 4);
            e    = (kk & 7) + (((kk >> 4) & 1) << 3);
        } else {      // B 32x16 tile: lanes 0-15 hold K{0..7,16..23}, lanes 16-31 hold K{8..15,24..31} (col N = L16)
            lane = L16 + (((kk >> 3) & 1) << 4);
            e    = (kk & 7) + (((kk >> 4) & 1) << 3);
        }
        const size_t off = ((size_t)(tile_r * KT_COUNT + kt) * 32 + lane) * 16 + e;
        hi_out[off] = h;
        lo_out[off] = lo;
    }
}

// ---------------- kernel: S_pos[i] = <f_norm[i], c_norm[labels[i]]> (exact f32) ----
__global__ __launch_bounds__(256)
void gpp_pos_kernel(const float* __restrict__ f, const float* __restrict__ c,
                    const int* __restrict__ labels,
                    const float* __restrict__ invf, const float* __restrict__ invc,
                    float* __restrict__ pos_s) {
    __shared__ float red[256];
    const int r = blockIdx.x;
    const int t = threadIdx.x;
    const int lab = labels[r];
    const float* fr = f + (size_t)r   * DIM;
    const float* cr = c + (size_t)lab * DIM;
    red[t] = fr[t] * cr[t] + fr[t + 256] * cr[t + 256];
    __syncthreads();
    for (int s = 128; s > 0; s >>= 1) {
        if (t < s) red[t] += red[t + s];
        __syncthreads();
    }
    if (t == 0) pos_s[r] = red[0] * invf[r] * invc[lab];
}

// ---------------- kernel: WMMA GEMM + fused softplus epilogue ----------------
// Block = 256 threads = 8 waves, wave grid 2x4; wave tile 32x64 -> block tile 64x256.
// Per k-step per wave: 12 fragment loads (384 B/lane), 24 WMMAs -> ~32.7 flop/byte.
__global__ __launch_bounds__(256)
void gpp_gemm_kernel(const unsigned short* __restrict__ Ah,
                     const unsigned short* __restrict__ Al,
                     const unsigned short* __restrict__ Bh,
                     const unsigned short* __restrict__ Bl,
                     const float* __restrict__ pos_s,
                     const int*   __restrict__ labels,
                     double* __restrict__ loss_acc,
                     unsigned int* __restrict__ cnt_acc) {
    const int lane = threadIdx.x & 31;
    const int w    = threadIdx.x >> 5;
    const int wr   = w >> 2;            // 0..1
    const int wc   = w & 3;             // 0..3
    const int mt0  = blockIdx.y * 4  + wr * 2;   // first of 2 consecutive 16-row tiles
    const int nt0  = blockIdx.x * 16 + wc * 4;   // first of 4 consecutive 16-col tiles

    v8f acc[2][4] = {};
    const size_t laneoff = (size_t)lane * 16;   // 16 bf16 = 32 bytes per lane

#pragma unroll 2
    for (int kt = 0; kt < KT_COUNT; ++kt) {
        v16bf ah[2], al[2], bh[4], bl[4];
#pragma unroll
        for (int i = 0; i < 2; ++i) {
            const size_t ta = ((size_t)(mt0 + i) * KT_COUNT + kt) * 512 + laneoff;
            ah[i] = *(const v16bf*)(Ah + ta);
            al[i] = *(const v16bf*)(Al + ta);
        }
#pragma unroll
        for (int i = 0; i < 4; ++i) {
            const size_t tb = ((size_t)(nt0 + i) * KT_COUNT + kt) * 512 + laneoff;
            bh[i] = *(const v16bf*)(Bh + tb);
            bl[i] = *(const v16bf*)(Bl + tb);
        }
#pragma unroll
        for (int mi = 0; mi < 2; ++mi) {
#pragma unroll
            for (int ni = 0; ni < 4; ++ni) {
                // bf16x3 fp32 emulation: hi*hi + hi*lo + lo*hi
                acc[mi][ni] = __builtin_amdgcn_wmma_f32_16x16x32_bf16(
                    false, ah[mi], false, bh[ni], (short)0, acc[mi][ni], false, false);
                acc[mi][ni] = __builtin_amdgcn_wmma_f32_16x16x32_bf16(
                    false, ah[mi], false, bl[ni], (short)0, acc[mi][ni], false, false);
                acc[mi][ni] = __builtin_amdgcn_wmma_f32_16x16x32_bf16(
                    false, al[mi], false, bh[ni], (short)0, acc[mi][ni], false, false);
            }
        }
    }

    // Epilogue: losses = softplus(S - S_pos), mask col != label, accumulate.
    float        lsum = 0.0f;
    unsigned int lcnt = 0u;
    const int half8 = (lane >> 4) << 3;   // C/D layout: lanes 16-31 hold rows +8
#pragma unroll
    for (int mi = 0; mi < 2; ++mi) {
#pragma unroll
        for (int ni = 0; ni < 4; ++ni) {
            const int col = (nt0 + ni) * 16 + (lane & 15);
#pragma unroll
            for (int g = 0; g < 8; ++g) {
                const int row = (mt0 + mi) * 16 + g + half8;
                const float x = acc[mi][ni][g] - pos_s[row];
                const float l = log1pf(expf(x));
                const bool valid = (col != labels[row]) && (l > 0.0f);
                lsum += valid ? l : 0.0f;
                lcnt += valid ? 1u : 0u;
            }
        }
    }

    __shared__ float        ssum[256];
    __shared__ unsigned int scnt[256];
    ssum[threadIdx.x] = lsum;
    scnt[threadIdx.x] = lcnt;
    __syncthreads();
    for (int s = 128; s > 0; s >>= 1) {
        if (threadIdx.x < s) {
            ssum[threadIdx.x] += ssum[threadIdx.x + s];
            scnt[threadIdx.x] += scnt[threadIdx.x + s];
        }
        __syncthreads();
    }
    if (threadIdx.x == 0) {
        atomicAdd(loss_acc, (double)ssum[0]);
        atomicAdd(cnt_acc, scnt[0]);
    }
}

// ---------------- kernel: finalize ----------------
__global__ void gpp_finalize_kernel(const double* loss_acc, const unsigned int* cnt_acc,
                                    float* out) {
    const double s = *loss_acc;
    const unsigned int c = *cnt_acc;
    out[0] = (float)((c > 0u) ? (s / (double)c) : s);
}

// ---------------- host launcher ----------------
extern "C" void kernel_launch(void* const* d_in, const int* in_sizes, int n_in,
                              void* d_out, int out_size, void* d_ws, size_t ws_size,
                              hipStream_t stream) {
    const float* features = (const float*)d_in[0];   // [4096, 512]
    const float* centers  = (const float*)d_in[1];   // [8192, 512]
    const int*   labels   = (const int*)d_in[2];     // [4096]
    // d_in[3] = cam_ids (unused by reference)
    float* out = (float*)d_out;

    char* ws = (char*)d_ws;
    unsigned short* Ah   = (unsigned short*)(ws + OFF_AHI);
    unsigned short* Al   = (unsigned short*)(ws + OFF_ALO);
    unsigned short* Bh   = (unsigned short*)(ws + OFF_BHI);
    unsigned short* Bl   = (unsigned short*)(ws + OFF_BLO);
    float*          invf = (float*)(ws + OFF_INVF);
    float*          invc = (float*)(ws + OFF_INVC);
    float*          poss = (float*)(ws + OFF_POS);
    double*         lacc = (double*)(ws + OFF_ACC);
    unsigned int*   cacc = (unsigned int*)(ws + OFF_ACC + 8);

    gpp_init_kernel<<<1, 1, 0, stream>>>(lacc, cacc);

    gpp_norm_swizzle_kernel<1><<<N_ROWS,    256, 0, stream>>>(features, Ah, Al, invf);
    gpp_norm_swizzle_kernel<0><<<N_CENTERS, 256, 0, stream>>>(centers,  Bh, Bl, invc);

    gpp_pos_kernel<<<N_ROWS, 256, 0, stream>>>(features, centers, labels, invf, invc, poss);

    dim3 grid(N_CENTERS / 256, N_ROWS / 64);   // 32 x 64 blocks, 64x256 tile each
    gpp_gemm_kernel<<<grid, 256, 0, stream>>>(Ah, Al, Bh, Bl, poss, labels, lacc, cacc);

    gpp_finalize_kernel<<<1, 1, 0, stream>>>(lacc, cacc, out);
}
